// FusionDecoderLayer_87222195848143
// MI455X (gfx1250) — compile-verified, matrix operand order checked by reading, not run on hardware
//
#include <hip/hip_runtime.h>
#include <cstdint>
#include <cstddef>

// ---------------------------------------------------------------------------
// CDNA5 (gfx1250) wave32 WMMA types
// ---------------------------------------------------------------------------
typedef __attribute__((ext_vector_type(16))) _Float16 v16h;
typedef __attribute__((ext_vector_type(8)))  float    v8f;

union Frag16 { v16h v; _Float16 h[16]; };
union FragC  { v8f  v; float     f[8]; };

// Load a 16x32 f16 fragment (A-layout / B^T-layout) from row-major f32.
// p must already point at (row, k0 + hi*8). Per the CDNA5 ISA 16-bit A layout:
//   lanes 0-15 (hi=0): VGPR j holds K = (j>>2)*16 + (j&3)*2 (+1)
//   lanes 16-31(hi=1): same +8
// => per lane the needed K values are [k0+hi*8 .. +7] and [k0+16+hi*8 .. +7].
__device__ inline void load_nt_frag(const float* __restrict__ p, Frag16& f) {
  float4 a = *(const float4*)(p);
  float4 b = *(const float4*)(p + 4);
  float4 c = *(const float4*)(p + 16);
  float4 d = *(const float4*)(p + 20);
  f.h[0]  = (_Float16)a.x; f.h[1]  = (_Float16)a.y; f.h[2]  = (_Float16)a.z; f.h[3]  = (_Float16)a.w;
  f.h[4]  = (_Float16)b.x; f.h[5]  = (_Float16)b.y; f.h[6]  = (_Float16)b.z; f.h[7]  = (_Float16)b.w;
  f.h[8]  = (_Float16)c.x; f.h[9]  = (_Float16)c.y; f.h[10] = (_Float16)c.z; f.h[11] = (_Float16)c.w;
  f.h[12] = (_Float16)d.x; f.h[13] = (_Float16)d.y; f.h[14] = (_Float16)d.z; f.h[15] = (_Float16)d.w;
}

struct GemmP {
  const float* A; const float* B; const float* bias; float* C;
  int M, N, K, lda, ldb, ldc;
  long sA, sB, sC;     // outer-batch strides
  long hA, hB, hC;     // head strides
  int H;               // heads per outer batch (1 if unused)
  float scale;
  int relu;
};

// ---------------------------------------------------------------------------
// NT GEMM: C = scale*(A @ B^T) + bias [+relu], B is [N,K] row-major.
// Weights/keys are L2-resident (192MB L2), so fragments load straight from
// global; next K-tile is prefetched (global_prefetch_b8) to overlap WMMAs.
// 256 threads = 8 waves (2x4); each wave owns a 32x32 tile (2x2 WMMA).
// ---------------------------------------------------------------------------
__global__ __launch_bounds__(256) void k_gemm_nt(GemmP p) {
  const int z  = blockIdx.z;
  const int bo = z / p.H;
  const int ho = z - bo * p.H;
  const float* __restrict__ A  = p.A + (size_t)bo * p.sA + (size_t)ho * p.hA;
  const float* __restrict__ Bm = p.B + (size_t)bo * p.sB + (size_t)ho * p.hB;
  float* __restrict__ C        = p.C + (size_t)bo * p.sC + (size_t)ho * p.hC;

  const int lane = threadIdx.x & 31;
  const int wave = threadIdx.x >> 5;
  const int wm = wave >> 2, wn = wave & 3;
  const int mBase = blockIdx.y * 64 + wm * 32;
  const int nBase = blockIdx.x * 128 + wn * 32;
  if (mBase >= p.M || nBase >= p.N) return;   // wave-uniform: EXEC stays full
  const int l15 = lane & 15, hi = lane >> 4;

  FragC acc[2][2];
#pragma unroll
  for (int t = 0; t < 2; ++t)
#pragma unroll
    for (int u = 0; u < 2; ++u)
#pragma unroll
      for (int i = 0; i < 8; ++i) acc[t][u].f[i] = 0.f;

  for (int k0 = 0; k0 < p.K; k0 += 32) {
    if (k0 + 32 < p.K) {   // prefetch next K tile (activations stream from HBM)
      __builtin_prefetch(A  + (size_t)(mBase + l15) * p.lda + k0 + 32, 0, 1);
      __builtin_prefetch(Bm + (size_t)(nBase + l15) * p.ldb + k0 + 32, 0, 1);
    }
    Frag16 a[2], b[2];
#pragma unroll
    for (int t = 0; t < 2; ++t)
      load_nt_frag(A + (size_t)(mBase + t * 16 + l15) * p.lda + k0 + hi * 8, a[t]);
#pragma unroll
    for (int t = 0; t < 2; ++t)
      load_nt_frag(Bm + (size_t)(nBase + t * 16 + l15) * p.ldb + k0 + hi * 8, b[t]);
#pragma unroll
    for (int t = 0; t < 2; ++t)
#pragma unroll
      for (int u = 0; u < 2; ++u)
        acc[t][u].v = __builtin_amdgcn_wmma_f32_16x16x32_f16(
            false, a[t].v, false, b[u].v, (short)0, acc[t][u].v, false, false);
  }

#pragma unroll
  for (int t = 0; t < 2; ++t)
#pragma unroll
    for (int u = 0; u < 2; ++u) {
      const int n = nBase + u * 16 + l15;
      if (n < p.N) {
        const float bb = p.bias ? p.bias[n] : 0.f;
#pragma unroll
        for (int i = 0; i < 8; ++i) {
          const int m = mBase + t * 16 + 8 * hi + i;   // C layout: VGPR i -> M=i (+8 for hi lanes)
          float v = acc[t][u].f[i] * p.scale + bb;
          if (p.relu) v = fmaxf(v, 0.f);
          C[(size_t)m * p.ldc + n] = v;
        }
      }
    }
}

// ---------------------------------------------------------------------------
// NN GEMM: C = scale*(A @ B) + bias, B is [K,N] row-major (P@V, g_attn@tailW).
// B columns are strided, so stage the 32x128 B tile to LDS transposed:
//   - each thread loads 16 contiguous floats (coalesced global_load_b128 x4)
//   - stores them n-major, row stride 40 halves (80B): 16-lane fragment reads
//     map to 16 distinct LDS banks, each fragment = two 16B ds_load_b128.
// No early return (block barriers); dead waves compute, stores are guarded.
// ---------------------------------------------------------------------------
__global__ __launch_bounds__(256) void k_gemm_nn(GemmP p) {
  __shared__ __align__(16) _Float16 sBT[128][40];
  const int z  = blockIdx.z;
  const int bo = z / p.H;
  const int ho = z - bo * p.H;
  const float* __restrict__ A  = p.A + (size_t)bo * p.sA + (size_t)ho * p.hA;
  const float* __restrict__ Bm = p.B + (size_t)bo * p.sB + (size_t)ho * p.hB;
  float* __restrict__ C        = p.C + (size_t)bo * p.sC + (size_t)ho * p.hC;

  const int lane = threadIdx.x & 31;
  const int wave = threadIdx.x >> 5;
  const int wm = wave >> 2, wn = wave & 3;
  const int mBase  = blockIdx.y * 64 + wm * 32;
  const int nBase0 = blockIdx.x * 128;
  const int nBase  = nBase0 + wn * 32;
  const int l15 = lane & 15, hi = lane >> 4;

  // staging coordinates: thread -> (k row, 16-col group)
  const int kr = threadIdx.x >> 3;
  const int cg = (threadIdx.x & 7) * 16;
  const bool inb = (nBase0 + cg) < p.N;   // N is a multiple of 64 -> whole group in/out

  FragC acc[2][2];
#pragma unroll
  for (int t = 0; t < 2; ++t)
#pragma unroll
    for (int u = 0; u < 2; ++u)
#pragma unroll
      for (int i = 0; i < 8; ++i) acc[t][u].f[i] = 0.f;

  for (int k0 = 0; k0 < p.K; k0 += 32) {
    __syncthreads();
    {
      float4 x0 = {0,0,0,0}, x1 = {0,0,0,0}, x2 = {0,0,0,0}, x3 = {0,0,0,0};
      if (inb) {
        const float* bp = Bm + (size_t)(k0 + kr) * p.ldb + nBase0 + cg;
        x0 = *(const float4*)(bp);
        x1 = *(const float4*)(bp + 4);
        x2 = *(const float4*)(bp + 8);
        x3 = *(const float4*)(bp + 12);
      }
      _Float16 hv[16];
      hv[0]=(_Float16)x0.x; hv[1]=(_Float16)x0.y; hv[2]=(_Float16)x0.z; hv[3]=(_Float16)x0.w;
      hv[4]=(_Float16)x1.x; hv[5]=(_Float16)x1.y; hv[6]=(_Float16)x1.z; hv[7]=(_Float16)x1.w;
      hv[8]=(_Float16)x2.x; hv[9]=(_Float16)x2.y; hv[10]=(_Float16)x2.z; hv[11]=(_Float16)x2.w;
      hv[12]=(_Float16)x3.x; hv[13]=(_Float16)x3.y; hv[14]=(_Float16)x3.z; hv[15]=(_Float16)x3.w;
#pragma unroll
      for (int c = 0; c < 16; ++c) sBT[cg + c][kr] = hv[c];   // transposed store
    }
    __syncthreads();

    if (k0 + 32 < p.K)
      __builtin_prefetch(A + (size_t)(mBase + l15) * p.lda + k0 + 32, 0, 1);

    Frag16 a[2], b[2];
#pragma unroll
    for (int t = 0; t < 2; ++t)
      load_nt_frag(A + (size_t)(mBase + t * 16 + l15) * p.lda + k0 + hi * 8, a[t]);
#pragma unroll
    for (int t = 0; t < 2; ++t) {
      const int nl = wn * 32 + t * 16 + l15;               // local column in tile
      *(float4*)&b[t].h[0] = *(const float4*)&sBT[nl][hi * 8];        // k = hi*8..+7
      *(float4*)&b[t].h[8] = *(const float4*)&sBT[nl][16 + hi * 8];   // k = 16+hi*8..+7
    }
#pragma unroll
    for (int t = 0; t < 2; ++t)
#pragma unroll
      for (int u = 0; u < 2; ++u)
        acc[t][u].v = __builtin_amdgcn_wmma_f32_16x16x32_f16(
            false, a[t].v, false, b[u].v, (short)0, acc[t][u].v, false, false);
  }

#pragma unroll
  for (int t = 0; t < 2; ++t)
#pragma unroll
    for (int u = 0; u < 2; ++u) {
      const int n = nBase + u * 16 + l15;
      if (n < p.N && mBase < p.M) {
        const float bb = p.bias ? p.bias[n] : 0.f;
#pragma unroll
        for (int i = 0; i < 8; ++i) {
          const int m = mBase + t * 16 + 8 * hi + i;
          float v = acc[t][u].f[i] * p.scale + bb;
          if (p.relu) v = fmaxf(v, 0.f);
          C[(size_t)m * p.ldc + n] = v;
        }
      }
    }
}

// ---------------------------------------------------------------------------
// Masked row softmax over scores [rows = gz*Lq, Lk]; causal and/or key-pad.
// ---------------------------------------------------------------------------
__global__ __launch_bounds__(256) void k_softmax(float* __restrict__ S, int Lq, int Lk, int H,
                                                 const int* __restrict__ padi,
                                                 const unsigned char* __restrict__ padb,
                                                 int causal) {
  __shared__ float red[256];
  const int r = blockIdx.x;
  const int q = r % Lq;
  const int b = (r / Lq) / H;
  float* row = S + (size_t)r * Lk;
  const int tid = threadIdx.x;

  float mx = -INFINITY;
  for (int k = tid; k < Lk; k += 256) {
    bool valid = (!causal) || (k <= q);
    if (padi) valid = valid && (padi[(size_t)b * Lk + k] != 0);
    if (padb) valid = valid && (padb[(size_t)b * Lk + k] == 0);
    float v = valid ? row[k] : -INFINITY;
    row[k] = v;
    mx = fmaxf(mx, v);
  }
  red[tid] = mx; __syncthreads();
  for (int o = 128; o; o >>= 1) { if (tid < o) red[tid] = fmaxf(red[tid], red[tid + o]); __syncthreads(); }
  mx = red[0]; __syncthreads();

  float sm = 0.f;
  for (int k = tid; k < Lk; k += 256) sm += __expf(row[k] - mx);
  red[tid] = sm; __syncthreads();
  for (int o = 128; o; o >>= 1) { if (tid < o) red[tid] += red[tid + o]; __syncthreads(); }
  const float inv = 1.f / red[0];
  for (int k = tid; k < Lk; k += 256) row[k] = __expf(row[k] - mx) * inv;
}

// mean over H=8 heads: out[b,q,k] = (1/8) * sum_h S[b,h,q,k]
__global__ void k_headmean(const float* __restrict__ S, float* __restrict__ out,
                           int Lq, int Lk, long n) {
  const long i = (long)blockIdx.x * 256 + threadIdx.x;
  if (i >= n) return;
  const long k = i % Lk;
  const long t = i / Lk;
  const long q = t % Lq;
  const long b = t / Lq;
  float s = 0.f;
#pragma unroll
  for (int h = 0; h < 8; ++h) s += S[(((b * 8 + h) * Lq + q)) * (long)Lk + k];
  out[i] = s * 0.125f;
}

// out = LayerNorm(a + b [+ c]) over E=512, gain g / bias be.
__global__ __launch_bounds__(128) void k_addln(const float* __restrict__ a,
                                               const float* __restrict__ b2,
                                               const float* __restrict__ c,
                                               const float* __restrict__ g,
                                               const float* __restrict__ be,
                                               float* __restrict__ out) {
  __shared__ float red[128];
  const size_t r = blockIdx.x;
  const int tid = threadIdx.x;
  const float* ap = a + r * 512;
  const float* bp = b2 + r * 512;
  const float* cp = c ? (c + r * 512) : nullptr;
  float v[4];
#pragma unroll
  for (int j = 0; j < 4; ++j) {
    const int e = tid + j * 128;
    v[j] = ap[e] + bp[e] + (cp ? cp[e] : 0.f);
  }
  red[tid] = v[0] + v[1] + v[2] + v[3]; __syncthreads();
  for (int o = 64; o; o >>= 1) { if (tid < o) red[tid] += red[tid + o]; __syncthreads(); }
  const float mu = red[0] * (1.f / 512.f);
  __syncthreads();
  float d = 0.f;
#pragma unroll
  for (int j = 0; j < 4; ++j) { const float t = v[j] - mu; d += t * t; }
  red[tid] = d; __syncthreads();
  for (int o = 64; o; o >>= 1) { if (tid < o) red[tid] += red[tid + o]; __syncthreads(); }
  const float inv = rsqrtf(red[0] * (1.f / 512.f) + 1e-5f);
#pragma unroll
  for (int j = 0; j < 4; ++j) {
    const int e = tid + j * 128;
    out[r * 512 + e] = (v[j] - mu) * inv * g[e] + be[e];
  }
}

// ---------------------------------------------------------------------------
// Fused graph attention: per (b, 16 query rows), loop 8 heads.
// WMMA scores for head/rel/tail (16x1024 each, f16 in LDS = 96KB),
// per-row LSE -> geometric-mean combine -> mask -> second softmax ->
// accumulate g_attn/H straight into d_out. Avoids 3x128MB of HBM traffic.
// ---------------------------------------------------------------------------
__global__ __launch_bounds__(256) void k_graph(
    const float* __restrict__ Qh, const float* __restrict__ Qr, const float* __restrict__ Qt,
    const float* __restrict__ Kh, const float* __restrict__ Kr, const float* __restrict__ Kt,
    const int* __restrict__ gmask, float* __restrict__ gOut) {
  __shared__ _Float16 ssc[3][16][1024];
  __shared__ float lse[3][16];
  const int b = blockIdx.y;
  const int qt = blockIdx.x;
  const int lane = threadIdx.x & 31;
  const int wave = threadIdx.x >> 5;
  const int l15 = lane & 15, hi = lane >> 4;
  const float* Qs[3] = {Qh, Qr, Qt};
  const float* Ks[3] = {Kh, Kr, Kt};

  for (int h = 0; h < 8; ++h) {
    // 1) score tiles: 8 waves cover the 64 key tiles
    for (int kt = wave; kt < 64; kt += 8) {
#pragma unroll
      for (int t = 0; t < 3; ++t) {
        FragC acc;
#pragma unroll
        for (int i = 0; i < 8; ++i) acc.f[i] = 0.f;
#pragma unroll
        for (int k0 = 0; k0 < 64; k0 += 32) {
          Frag16 af, bf;
          load_nt_frag(Qs[t] + (size_t)(b * 512 + qt * 16 + l15) * 512 + h * 64 + k0 + hi * 8, af);
          load_nt_frag(Ks[t] + (size_t)(b * 1024 + kt * 16 + l15) * 512 + h * 64 + k0 + hi * 8, bf);
          acc.v = __builtin_amdgcn_wmma_f32_16x16x32_f16(false, af.v, false, bf.v,
                                                         (short)0, acc.v, false, false);
        }
#pragma unroll
        for (int i = 0; i < 8; ++i)
          ssc[t][i + 8 * hi][kt * 16 + l15] = (_Float16)(acc.f[i] * 0.125f);
      }
    }
    __syncthreads();

    // 2) per-row LSE for each of 3 score sets (48 reductions / 8 waves)
    for (int idx = wave * 6; idx < wave * 6 + 6; ++idx) {
      const int t = idx >> 4;
      const int r = idx & 15;
      float mx = -INFINITY;
      for (int k = lane; k < 1024; k += 32) mx = fmaxf(mx, (float)ssc[t][r][k]);
      for (int o = 16; o; o >>= 1) mx = fmaxf(mx, __shfl_xor(mx, o, 32));
      float sm = 0.f;
      for (int k = lane; k < 1024; k += 32) sm += __expf((float)ssc[t][r][k] - mx);
      for (int o = 16; o; o >>= 1) sm += __shfl_xor(sm, o, 32);
      if (lane == 0) lse[t][r] = mx + __logf(sm);
    }
    __syncthreads();

    // 3) gw = softmax(mask(exp(mean log p))); accumulate /H into gOut
    for (int rr = 0; rr < 2; ++rr) {
      const int r = wave * 2 + rr;
      const float l3 = (lse[0][r] + lse[1][r] + lse[2][r]) * (1.f / 3.f);
      float mx = -INFINITY;
      for (int k = lane; k < 1024; k += 32) {
        const float s3 = ((float)ssc[0][r][k] + (float)ssc[1][r][k] + (float)ssc[2][r][k]) * (1.f / 3.f);
        const float gp = (gmask[b * 1024 + k] != 0) ? __expf(s3 - l3) : -INFINITY;
        ssc[0][r][k] = (_Float16)gp;   // row owned exclusively by this wave
        mx = fmaxf(mx, gp);
      }
      for (int o = 16; o; o >>= 1) mx = fmaxf(mx, __shfl_xor(mx, o, 32));
      float sm = 0.f;
      for (int k = lane; k < 1024; k += 32) sm += __expf((float)ssc[0][r][k] - mx);
      for (int o = 16; o; o >>= 1) sm += __shfl_xor(sm, o, 32);
      const float inv = 0.125f / sm;   // /H folded in
      float* go = gOut + ((size_t)b * 512 + qt * 16 + r) * 1024;
      for (int k = lane; k < 1024; k += 32) {
        const float v = __expf((float)ssc[0][r][k] - mx) * inv;
        if (h == 0) go[k] = v; else go[k] += v;
      }
    }
    __syncthreads();
  }
}

// ---------------------------------------------------------------------------
// Host-side orchestration
// ---------------------------------------------------------------------------
static inline void gemm(hipStream_t st, const float* A, const float* B, const float* bias,
                        float* C, int M, int N, int K, int lda, int ldb, int ldc,
                        long sA, long sB, long sC, long hA, long hB, long hC,
                        int gz, int H, float scale, bool transB, bool relu) {
  GemmP p;
  p.A = A; p.B = B; p.bias = bias; p.C = C;
  p.M = M; p.N = N; p.K = K; p.lda = lda; p.ldb = ldb; p.ldc = ldc;
  p.sA = sA; p.sB = sB; p.sC = sC; p.hA = hA; p.hB = hB; p.hC = hC;
  p.H = H; p.scale = scale; p.relu = relu ? 1 : 0;
  dim3 g((N + 127) / 128, (M + 63) / 64, gz);
  if (transB) k_gemm_nt<<<g, 256, 0, st>>>(p);
  else        k_gemm_nn<<<g, 256, 0, st>>>(p);
}

extern "C" void kernel_launch(void* const* d_in, const int* in_sizes, int n_in,
                              void* d_out, int out_size, void* d_ws, size_t ws_size,
                              hipStream_t stream) {
  (void)in_sizes; (void)n_in; (void)out_size; (void)ws_size;

  const float* memory = (const float*)d_in[0];
  const float* reply  = (const float*)d_in[1];
  const float* doc    = (const float*)d_in[2];
  const unsigned char* mem_mask = (const unsigned char*)d_in[3];  // bool: True = masked
  const int* reply_mask = (const int*)d_in[4];                    // 1 = valid
  const int* doc_mask   = (const int*)d_in[5];
  const float* head = (const float*)d_in[6];
  const float* rel  = (const float*)d_in[7];
  const float* tail = (const float*)d_in[8];
  const int* gmask  = (const int*)d_in[9];
  const float* s_iw = (const float*)d_in[10]; const float* s_ib = (const float*)d_in[11];
  const float* s_ow = (const float*)d_in[12]; const float* s_ob = (const float*)d_in[13];
  const float* c_iw = (const float*)d_in[14]; const float* c_ib = (const float*)d_in[15];
  const float* c_ow = (const float*)d_in[16]; const float* c_ob = (const float*)d_in[17];
  const float* d_iw = (const float*)d_in[18]; const float* d_ib = (const float*)d_in[19];
  const float* d_ow = (const float*)d_in[20]; const float* d_ob = (const float*)d_in[21];
  const float* h_iw = (const float*)d_in[22]; const float* h_ib = (const float*)d_in[23];
  const float* r_iw = (const float*)d_in[26]; const float* r_ib = (const float*)d_in[27];
  const float* t_iw = (const float*)d_in[30]; const float* t_ib = (const float*)d_in[31];
  const float* ln1g = (const float*)d_in[34]; const float* ln1b = (const float*)d_in[35];
  const float* ln2g = (const float*)d_in[36]; const float* ln2b = (const float*)d_in[37];
  const float* ln3g = (const float*)d_in[38]; const float* ln3b = (const float*)d_in[39];
  const float* ln4g = (const float*)d_in[40]; const float* ln4b = (const float*)d_in[41];
  const float* W_w  = (const float*)d_in[42];
  const float* w1 = (const float*)d_in[43]; const float* b1 = (const float*)d_in[44];
  const float* w2 = (const float*)d_in[45]; const float* b2 = (const float*)d_in[46];

  // Output layout: x | h_attn | g_attn | d_attn
  float* xOut = (float*)d_out;
  float* hOut = xOut + 2097152;   // [8,512,512]
  float* gOut = hOut + 2097152;   // [8,512,1024]
  float* dOut = gOut + 4194304;   // [8,512,1024]

  // Workspace layout (floats)
  float* W = (float*)d_ws;
  float* Qp   = W;                  // 2,097,152  [B*512, 512]
  float* Kp   = Qp   + 2097152;     // 4,194,304  [B*1024, 512] max
  float* Vp   = Kp   + 4194304;     // 4,194,304
  float* Sp   = Vp   + 4194304;     // 33,554,432 [B,H,512,1024] max
  float* CTXp = Sp   + 33554432;    // 2,097,152
  float* AOp  = CTXp + 2097152;     // 2,097,152
  float* X1p  = AOp  + 2097152;     // 2,097,152
  float* X2p  = X1p  + 2097152;     // 2,097,152
  float* XGp  = X2p  + 2097152;     // 2,097,152
  float* TWp  = XGp  + 2097152;     // 4,194,304
  float* QHp  = TWp  + 4194304;     float* QRp = QHp + 2097152; float* QTp = QRp + 2097152;
  float* KHp  = QTp  + 2097152;     float* KRp = KHp + 4194304; float* KTp = KRp + 4194304;
  float* HHp  = KTp  + 4194304;     // 8,388,608 [4096,2048]

  const long LqE = 512L * 512, LkE = 512L * 512, LgE = 1024L * 512;

  // ---- self attention (causal + key-pad) ----
  gemm(stream, reply, s_iw,               s_ib,        Qp, 4096, 512, 512, 512, 512, 512, 0,0,0,0,0,0, 1,1, 1.f, true, false);
  gemm(stream, reply, s_iw + 512*512,     s_ib + 512,  Kp, 4096, 512, 512, 512, 512, 512, 0,0,0,0,0,0, 1,1, 1.f, true, false);
  gemm(stream, reply, s_iw + 2*512*512,   s_ib + 1024, Vp, 4096, 512, 512, 512, 512, 512, 0,0,0,0,0,0, 1,1, 1.f, true, false);
  gemm(stream, Qp, Kp, nullptr, Sp, 512, 512, 64, 512, 512, 512,
       LqE, LkE, 8L*512*512, 64, 64, 512L*512, 64, 8, 0.125f, true, false);
  k_softmax<<<32768, 256, 0, stream>>>(Sp, 512, 512, 8, reply_mask, nullptr, 1);
  gemm(stream, Sp, Vp, nullptr, CTXp, 512, 64, 512, 512, 512, 512,
       8L*512*512, LkE, LqE, 512L*512, 64, 64, 64, 8, 1.f, false, false);
  gemm(stream, CTXp, s_ow, s_ob, AOp, 4096, 512, 512, 512, 512, 512, 0,0,0,0,0,0, 1,1, 1.f, true, false);
  k_addln<<<4096, 128, 0, stream>>>(reply, AOp, nullptr, ln1g, ln1b, X1p);

  // ---- cross attention (query = ORIGINAL reply) ----
  gemm(stream, reply,  c_iw,             c_ib,        Qp, 4096, 512, 512, 512, 512, 512, 0,0,0,0,0,0, 1,1, 1.f, true, false);
  gemm(stream, memory, c_iw + 512*512,   c_ib + 512,  Kp, 4096, 512, 512, 512, 512, 512, 0,0,0,0,0,0, 1,1, 1.f, true, false);
  gemm(stream, memory, c_iw + 2*512*512, c_ib + 1024, Vp, 4096, 512, 512, 512, 512, 512, 0,0,0,0,0,0, 1,1, 1.f, true, false);
  gemm(stream, Qp, Kp, nullptr, Sp, 512, 512, 64, 512, 512, 512,
       LqE, LkE, 8L*512*512, 64, 64, 512L*512, 64, 8, 0.125f, true, false);
  k_softmax<<<32768, 256, 0, stream>>>(Sp, 512, 512, 8, nullptr, mem_mask, 0);
  k_headmean<<<(2097152 + 255) / 256, 256, 0, stream>>>(Sp, hOut, 512, 512, 2097152L);
  gemm(stream, Sp, Vp, nullptr, CTXp, 512, 64, 512, 512, 512, 512,
       8L*512*512, LkE, LqE, 512L*512, 64, 64, 64, 8, 1.f, false, false);
  gemm(stream, CTXp, c_ow, c_ob, AOp, 4096, 512, 512, 512, 512, 512, 0,0,0,0,0,0, 1,1, 1.f, true, false);
  k_addln<<<4096, 128, 0, stream>>>(X1p, AOp, nullptr, ln2g, ln2b, X2p);

  // ---- graph attention (fused 3-way geometric-mean combine) ----
  gemm(stream, X2p,  h_iw,           h_ib,       QHp, 4096, 512, 512, 512, 512, 512, 0,0,0,0,0,0, 1,1, 1.f, true, false);
  gemm(stream, head, h_iw + 512*512, h_ib + 512, KHp, 8192, 512, 512, 512, 512, 512, 0,0,0,0,0,0, 1,1, 1.f, true, false);
  gemm(stream, X2p,  r_iw,           r_ib,       QRp, 4096, 512, 512, 512, 512, 512, 0,0,0,0,0,0, 1,1, 1.f, true, false);
  gemm(stream, rel,  r_iw + 512*512, r_ib + 512, KRp, 8192, 512, 512, 512, 512, 512, 0,0,0,0,0,0, 1,1, 1.f, true, false);
  gemm(stream, X2p,  t_iw,           t_ib,       QTp, 4096, 512, 512, 512, 512, 512, 0,0,0,0,0,0, 1,1, 1.f, true, false);
  gemm(stream, tail, t_iw + 512*512, t_ib + 512, KTp, 8192, 512, 512, 512, 512, 512, 0,0,0,0,0,0, 1,1, 1.f, true, false);
  k_graph<<<dim3(32, 8), 256, 0, stream>>>(QHp, QRp, QTp, KHp, KRp, KTp, gmask, gOut);
  gemm(stream, tail, W_w, nullptr, TWp, 8192, 512, 512, 512, 512, 512, 0,0,0,0,0,0, 1,1, 1.f, true, false);
  gemm(stream, gOut, TWp, nullptr, XGp, 512, 512, 1024, 1024, 512, 512,
       512L*1024, LgE, LqE, 0, 0, 0, 8, 1, 1.f, false, false);

  // ---- doc attention ----
  gemm(stream, X2p, d_iw,             d_ib,        Qp, 4096, 512, 512, 512, 512, 512, 0,0,0,0,0,0, 1,1, 1.f, true, false);
  gemm(stream, doc, d_iw + 512*512,   d_ib + 512,  Kp, 8192, 512, 512, 512, 512, 512, 0,0,0,0,0,0, 1,1, 1.f, true, false);
  gemm(stream, doc, d_iw + 2*512*512, d_ib + 1024, Vp, 8192, 512, 512, 512, 512, 512, 0,0,0,0,0,0, 1,1, 1.f, true, false);
  gemm(stream, Qp, Kp, nullptr, Sp, 512, 1024, 64, 512, 512, 1024,
       LqE, LgE, 8L*512*1024, 64, 64, 512L*1024, 64, 8, 0.125f, true, false);
  k_softmax<<<32768, 256, 0, stream>>>(Sp, 512, 1024, 8, doc_mask, nullptr, 0);
  k_headmean<<<(4194304 + 255) / 256, 256, 0, stream>>>(Sp, dOut, 512, 1024, 4194304L);
  gemm(stream, Sp, Vp, nullptr, CTXp, 512, 64, 1024, 1024, 512, 512,
       8L*512*1024, LgE, LqE, 512L*1024, 64, 64, 64, 8, 1.f, false, false);
  gemm(stream, CTXp, d_ow, d_ob, AOp, 4096, 512, 512, 512, 512, 512, 0,0,0,0,0,0, 1,1, 1.f, true, false);
  k_addln<<<4096, 128, 0, stream>>>(X2p, AOp, XGp, ln3g, ln3b, X1p);

  // ---- FFN ----
  gemm(stream, X1p, w1, b1, HHp, 4096, 2048, 512, 512, 512, 2048, 0,0,0,0,0,0, 1,1, 1.f, true, true);
  gemm(stream, HHp, w2, b2, AOp, 4096, 512, 2048, 2048, 2048, 512, 0,0,0,0,0,0, 1,1, 1.f, true, false);
  k_addln<<<4096, 128, 0, stream>>>(X1p, AOp, nullptr, ln4g, ln4b, xOut);
}
